// FixedWeightDense_63831803953664
// MI455X (gfx1250) — compile-verified, matrix operand order checked by reading, not run on hardware
//
#include <hip/hip_runtime.h>

// MI455X (gfx1250) batched GEMM, fp32 in/out, bf16x3 WMMA emulation.
// Pass 1: one-time fp32 -> bf16 hi/lo split into workspace (B transposed to [n][k]).
// Pass 2: WMMA GEMM with double-buffered LDS fed by async global->LDS copies.

typedef __attribute__((ext_vector_type(16))) __bf16 v16bf;
typedef __attribute__((ext_vector_type(8)))  float  v8f;
typedef unsigned int u32;
typedef unsigned short u16;
typedef __attribute__((ext_vector_type(2))) u32 u32x2;
typedef __attribute__((ext_vector_type(4))) u32 u32x4;
typedef __attribute__((ext_vector_type(4))) int i32x4;
typedef __attribute__((address_space(1))) i32x4 gas_i32x4;   // global
typedef __attribute__((address_space(3))) i32x4 las_i32x4;   // LDS

#define S_DIM 2048
#define K_DIM 2048
#define N_DIM 2048
#define BM 128
#define BN 128
#define BK 32
#define NTHREADS 256           // 8 wave32
#define NT (K_DIM / BK)

#if __has_builtin(__builtin_amdgcn_global_load_async_to_lds_b128)
#define USE_ASYNC 1
#else
#define USE_ASYNC 0
#endif

__device__ __forceinline__ u16 bf16_rne(float x) {
  u32 u = __float_as_uint(x);
  u32 r = u + 0x7FFFu + ((u >> 16) & 1u);
  return (u16)(r >> 16);
}
__device__ __forceinline__ float bf16_f(u16 h) {
  return __uint_as_float(((u32)h) << 16);
}
__device__ __forceinline__ u32 pack2(u16 lo, u16 hi) {
  return (u32)lo | ((u32)hi << 16);
}

// ---------------- Pass 1a: elementwise split of A ----------------
__global__ __launch_bounds__(NTHREADS)
void split_a(const float* __restrict__ in, u16* __restrict__ hi,
             u16* __restrict__ lo, long n4) {
  long i = (long)blockIdx.x * NTHREADS + threadIdx.x;
  if (i >= n4) return;
  float4 v = ((const float4*)in)[i];
  float f[4] = {v.x, v.y, v.z, v.w};
  u16 h[4], l[4];
#pragma unroll
  for (int j = 0; j < 4; ++j) {
    h[j] = bf16_rne(f[j]);
    l[j] = bf16_rne(f[j] - bf16_f(h[j]));
  }
  ((u32x2*)hi)[i] = (u32x2){pack2(h[0], h[1]), pack2(h[2], h[3])};
  ((u32x2*)lo)[i] = (u32x2){pack2(l[0], l[1]), pack2(l[2], l[3])};
}

// ------- Pass 1b: split of B with 32x32 LDS-tiled transpose to [b][n][k] -------
__global__ __launch_bounds__(NTHREADS)
void split_b_t(const float* __restrict__ in, u16* __restrict__ hi,
               u16* __restrict__ lo) {
  __shared__ u16 tH[32][34];
  __shared__ u16 tL[32][34];
  const int bz = blockIdx.z, k0 = blockIdx.y * 32, nn0 = blockIdx.x * 32;
  const int t = threadIdx.x, r = t >> 3, c4 = (t & 7) * 4;
  float4 v = *(const float4*)(in + ((size_t)bz * K_DIM + k0 + r) * (size_t)N_DIM + nn0 + c4);
  float f[4] = {v.x, v.y, v.z, v.w};
#pragma unroll
  for (int j = 0; j < 4; ++j) {
    u16 h = bf16_rne(f[j]);
    tH[r][c4 + j] = h;
    tL[r][c4 + j] = bf16_rne(f[j] - bf16_f(h));
  }
  __syncthreads();
  u16 oh[4], ol[4];
#pragma unroll
  for (int j = 0; j < 4; ++j) { oh[j] = tH[c4 + j][r]; ol[j] = tL[c4 + j][r]; }
  const size_t o = ((size_t)bz * N_DIM + nn0 + r) * (size_t)K_DIM + k0 + c4;
  *(u32x2*)(hi + o) = (u32x2){pack2(oh[0], oh[1]), pack2(oh[2], oh[3])};
  *(u32x2*)(lo + o) = (u32x2){pack2(ol[0], ol[1]), pack2(ol[2], ol[3])};
}

// ---------------- Pass 2: WMMA GEMM ----------------
// LDS rows hold 32 bf16 of K; 16B chunks are XOR-swizzled by ((row>>2)&3) so
// fragment b128 reads are bank-conflict-free without padding (64 KB total).
__device__ __forceinline__ v16bf ld_frag_sw(const u16* rowBase, int row, int kh8) {
  const int sw = (row >> 2) & 3;
  union { u32x4 q[2]; v16bf v; } u;
  u.q[0] = *(const u32x4*)(rowBase + 8 * ((kh8)     ^ sw));  // K kh..kh+7
  u.q[1] = *(const u32x4*)(rowBase + 8 * ((kh8 + 2) ^ sw));  // K 16+kh..
  return u.v;
}

__global__ __launch_bounds__(NTHREADS)
void bgemm_bf16x3_wmma(const u16* __restrict__ AH, const u16* __restrict__ AL,
                       const u16* __restrict__ BH, const u16* __restrict__ BL,
                       float* __restrict__ Out) {
  __shared__ __align__(16) u16 sm[2][4][BM * BK];   // 2 stages x {AH,AL,BH,BL}

  const int tid  = threadIdx.x;
  const int lane = tid & 31;
  const int wid  = tid >> 5;
  const int wm   = wid >> 2;          // 0..1
  const int wn   = wid & 3;           // 0..3
  const int bz   = blockIdx.z;
  const int m0   = blockIdx.y * BM;
  const int n0   = blockIdx.x * BN;

  // Staging: 128 rows x 4 chunks (16B) per buffer = 512 b128 / 256 thr = 2 each.
  int sRow[2], sChk[2];
#pragma unroll
  for (int i = 0; i < 2; ++i) {
    int idx = tid + i * NTHREADS;
    sRow[i] = idx >> 2;
    sChk[i] = idx & 3;
  }
  const u16* gsrc[4];
  gsrc[0] = AH + ((size_t)bz * S_DIM + m0) * (size_t)K_DIM;
  gsrc[1] = AL + ((size_t)bz * S_DIM + m0) * (size_t)K_DIM;
  gsrc[2] = BH + ((size_t)bz * N_DIM + n0) * (size_t)K_DIM;   // [n][k] layout
  gsrc[3] = BL + ((size_t)bz * N_DIM + n0) * (size_t)K_DIM;

  auto lds_off = [&](int i) -> int {
    return sRow[i] * BK + 8 * (sChk[i] ^ ((sRow[i] >> 2) & 3));
  };

#if USE_ASYNC
  auto issue = [&](int k0, int stage) {
#pragma unroll
    for (int b = 0; b < 4; ++b)
#pragma unroll
      for (int i = 0; i < 2; ++i) {
        const u16* g = gsrc[b] + (size_t)sRow[i] * K_DIM + k0 + sChk[i] * 8;
        __builtin_amdgcn_global_load_async_to_lds_b128(
            (gas_i32x4*)g,
            (las_i32x4*)&sm[stage][b][lds_off(i)],
            0, 0);
      }
  };
#else
  u32x4 regs[4][2];
  auto load_regs = [&](int k0) {
#pragma unroll
    for (int b = 0; b < 4; ++b)
#pragma unroll
      for (int i = 0; i < 2; ++i)
        regs[b][i] = *(const u32x4*)(gsrc[b] + (size_t)sRow[i] * K_DIM + k0 + sChk[i] * 8);
  };
  auto store_stage = [&](int stage) {
#pragma unroll
    for (int b = 0; b < 4; ++b)
#pragma unroll
      for (int i = 0; i < 2; ++i)
        *(u32x4*)&sm[stage][b][lds_off(i)] = regs[b][i];
  };
#endif

  v8f acc[4][2] = {};
  const int lr  = lane & 15;
  const int kh8 = lane >> 4;          // chunk pair {kh8, kh8+2} per ISA A-layout

  auto compute = [&](int stage) {
    v16bf fAH[4], fAL[4];
#pragma unroll
    for (int mt = 0; mt < 4; ++mt) {
      const int row = wm * 64 + mt * 16 + lr;
      fAH[mt] = ld_frag_sw(&sm[stage][0][row * BK], row, kh8);
      fAL[mt] = ld_frag_sw(&sm[stage][1][row * BK], row, kh8);
    }
    v16bf fBH[2], fBL[2];
#pragma unroll
    for (int nt = 0; nt < 2; ++nt) {
      const int row = wn * 32 + nt * 16 + lr;
      fBH[nt] = ld_frag_sw(&sm[stage][2][row * BK], row, kh8);
      fBL[nt] = ld_frag_sw(&sm[stage][3][row * BK], row, kh8);
    }
#pragma unroll
    for (int mt = 0; mt < 4; ++mt)
#pragma unroll
      for (int nt = 0; nt < 2; ++nt) {
        v8f c = acc[mt][nt];
        // bf16x3: hi*hi + hi*lo + lo*hi (lo*lo below fp32 ulp at K=2048)
        c = __builtin_amdgcn_wmma_f32_16x16x32_bf16(false, fAH[mt], false, fBH[nt], (short)0, c, false, false);
        c = __builtin_amdgcn_wmma_f32_16x16x32_bf16(false, fAH[mt], false, fBL[nt], (short)0, c, false, false);
        c = __builtin_amdgcn_wmma_f32_16x16x32_bf16(false, fAL[mt], false, fBH[nt], (short)0, c, false, false);
        acc[mt][nt] = c;
      }
  };

#if USE_ASYNC
  issue(0, 0);
  for (int kt = 0; kt < NT; ++kt) {
    const int cur = kt & 1;
    asm volatile("s_wait_asynccnt 0x0" ::: "memory");   // my stage-cur copies done
    __syncthreads();                                    // everyone's done; prev reads of cur^1 done
    if (kt + 1 < NT) issue((kt + 1) * BK, cur ^ 1);
    compute(cur);
  }
#else
  load_regs(0);
  for (int kt = 0; kt < NT; ++kt) {
    const int cur = kt & 1;
    store_stage(cur);
    __syncthreads();
    if (kt + 1 < NT) load_regs((kt + 1) * BK);
    compute(cur);
  }
#endif

  // C/D layout: VGPR r -> (lanes 0-15: M=r, N=lane) (lanes 16-31: M=r+8, N=lane-16)
  float* outBase = Out + ((size_t)bz * S_DIM + m0 + wm * 64) * (size_t)N_DIM + n0 + wn * 32;
  const int cn    = lane & 15;
  const int rmOff = (lane >> 4) * 8;
#pragma unroll
  for (int mt = 0; mt < 4; ++mt)
#pragma unroll
    for (int nt = 0; nt < 2; ++nt) {
      float* c = outBase + (size_t)(mt * 16 + rmOff) * N_DIM + nt * 16 + cn;
#pragma unroll
      for (int r = 0; r < 8; ++r) c[(size_t)r * N_DIM] = acc[mt][nt][r];
    }
}

extern "C" void kernel_launch(void* const* d_in, const int* in_sizes, int n_in,
                              void* d_out, int out_size, void* d_ws, size_t ws_size,
                              hipStream_t stream) {
  const float* a = (const float*)d_in[0];
  const float* b = (const float*)d_in[1];
  float* out = (float*)d_out;
  const int Bt = in_sizes[0] / (S_DIM * K_DIM);         // 8
  const size_t elems = (size_t)Bt * S_DIM * K_DIM;      // 33.5M

  // Workspace layout: AH | AL | BH | BL (bf16 each) = 4 * 67 MB = 268 MB.
  u16* AH = (u16*)d_ws;
  u16* AL = AH + elems;
  u16* BH = AL + elems;
  u16* BL = BH + elems;

  const long n4 = (long)(elems / 4);
  split_a<<<dim3((unsigned)((n4 + NTHREADS - 1) / NTHREADS)), dim3(NTHREADS), 0, stream>>>(a, AH, AL, n4);
  split_b_t<<<dim3(N_DIM / 32, K_DIM / 32, Bt), dim3(NTHREADS), 0, stream>>>(b, BH, BL);

  dim3 grid(N_DIM / BN, S_DIM / BM, Bt);                // 16 x 16 x 8
  bgemm_bf16x3_wmma<<<grid, dim3(NTHREADS), 0, stream>>>(AH, AL, BH, BL, out);
}
